// CRGCN_72619307041534
// MI455X (gfx1250) — matrix operand order repaired and checked by reading, not compiled, not run on hardware
//
#include <hip/hip_runtime.h>
#include <math.h>

#define N_USERS   100000
#define N_ITEMS   100000
#define N_NODES   200000
#define EMBED     64
#define NEDGE     2000000
#define BATCH_N   4096
#define REG_W     0.0001f

typedef __attribute__((ext_vector_type(16))) _Float16 v16h;
typedef __attribute__((ext_vector_type(8)))  float    v8f;

// ---------------- utility kernels ----------------

__global__ void k_zero_acc(float* acc) {
    if (threadIdx.x < 16) acc[threadIdx.x] = 0.f;
}

__global__ void k_init_total(const float* __restrict__ u, const float* __restrict__ it,
                             float* __restrict__ total) {
    size_t i = (size_t)blockIdx.x * blockDim.x + threadIdx.x;   // float4 index
    const size_t nU = (size_t)N_USERS * EMBED / 4;
    const size_t nT = (size_t)N_NODES * EMBED / 4;
    if (i >= nT) return;
    float4 v = (i < nU) ? ((const float4*)u)[i] : ((const float4*)it)[i - nU];
    ((float4*)total)[i] = v;
}

__global__ void k_zero(float* __restrict__ p, long long n4) {
    long long i = (long long)blockIdx.x * blockDim.x + threadIdx.x;
    long long stride = (long long)gridDim.x * blockDim.x;
    float4 z = make_float4(0.f, 0.f, 0.f, 0.f);
    for (; i < n4; i += stride) ((float4*)p)[i] = z;
}

// ---------------- WMMA GEMM: x = total @ W (f16 in, f32 acc) ----------------
// Pre-swizzle W[bi] (64x64 f32, row-major, K x N) into per-lane-contiguous
// B fragments: bp[((colTile*2 + g)*32 + lane)*16 + h] = f16(W[32g + 8*(lane>>4) + h + (h>=8 ? 8 : 0)][colTile*16 + (lane&15)])
__global__ void k_pack_w(const float* __restrict__ W, _Float16* __restrict__ bp) {
    int t = threadIdx.x;                 // 256 threads: (colTile*2+g)*32 + lane
    if (t >= 256) return;
    int lane = t & 31;
    int g = (t >> 5) & 1;
    int colTile = t >> 6;
    int N = colTile * 16 + (lane & 15);
    int kbase = 32 * g + 8 * (lane >> 4);
#pragma unroll
    for (int h = 0; h < 16; ++h)
        bp[t * 16 + h] = (_Float16)W[(kbase + h + (h >= 8 ? 8 : 0)) * EMBED + N];
}

// One wave per 16x16 output tile. rowTile in [0,12500), colTile in [0,4).
__global__ void k_gemm(const float* __restrict__ T, const _Float16* __restrict__ bp,
                       float* __restrict__ X) {
    int wave = ((int)blockIdx.x * blockDim.x + threadIdx.x) >> 5;
    int lane = threadIdx.x & 31;
    int rowTile = wave >> 2;
    int colTile = wave & 3;
    int m0 = rowTile * 16;
    int n0 = colTile * 16;
    int mlo = lane & 15;
    int hi  = lane >> 4;                 // 0 or 1
    const float* arow = T + (size_t)(m0 + mlo) * EMBED;

    v8f c = {};
#pragma unroll
    for (int g = 0; g < 2; ++g) {
        // A fragment per ISA 16-bit 16x32 layout:
        // halves 0..7  -> K = 32g + 8*hi + j
        // halves 8..15 -> K = 32g + 8*hi + 16 + j
        int b0 = 32 * g + 8 * hi;
        v16h a;
#pragma unroll
        for (int j = 0; j < 8; ++j) a[j]     = (_Float16)arow[b0 + j];
#pragma unroll
        for (int j = 0; j < 8; ++j) a[8 + j] = (_Float16)arow[b0 + 16 + j];
        v16h b = *(const v16h*)(bp + (((colTile * 2 + g) * 32 + lane) << 4));
        c = __builtin_amdgcn_wmma_f32_16x16x32_f16(false, a, false, b,
                                                   (short)0, c, false, false);
    }
    // C/D layout: VGPR j -> row m0 + j + 8*hi, col n0 + (lane&15)
    float* xp = X + (size_t)(m0 + 8 * hi) * EMBED + n0 + mlo;
#pragma unroll
    for (int j = 0; j < 8; ++j) xp[(size_t)j * EMBED] = c[j];
}

// ---------------- graph conv scatter phase ----------------

__global__ void k_deg(const int* __restrict__ cols, float* __restrict__ deg) {
    int e = (int)blockIdx.x * blockDim.x + threadIdx.x;
    if (e >= NEDGE) return;
    atomicAdd(&deg[cols[e]], 1.0f);
}

__global__ void k_dinv(const float* __restrict__ deg, float* __restrict__ dinv) {
    int i = (int)blockIdx.x * blockDim.x + threadIdx.x;
    if (i >= N_NODES) return;
    float d = deg[i];
    dinv[i] = (d > 0.f) ? rsqrtf(d) : 0.f;   // d>=1 when nonzero, so max(d,1)=d
}

// One wave per edge, 2 dims per lane.
__global__ void k_scatter(const int* __restrict__ rows, const int* __restrict__ cols,
                          const float* __restrict__ dinv, const float* __restrict__ x,
                          float* __restrict__ out) {
    long long gid = (long long)blockIdx.x * blockDim.x + threadIdx.x;
    int e = (int)(gid >> 5);
    int lane = (int)(gid & 31);
    if (e >= NEDGE) return;
    int r = rows[e], c = cols[e];
    float nrm = dinv[r] * dinv[c];
    if (nrm == 0.f) return;
    float2 v = *(const float2*)(x + (size_t)r * EMBED + 2 * lane);
    float* o = out + (size_t)c * EMBED + 2 * lane;
    atomicAdd(o,     nrm * v.x);
    atomicAdd(o + 1, nrm * v.y);
}

// layer = l2_normalize(out + bias); total += layer.  One wave per node.
__global__ void k_norm_add(const float* __restrict__ out, const float* __restrict__ bias,
                           float* __restrict__ total) {
    int wave = ((int)blockIdx.x * blockDim.x + threadIdx.x) >> 5;
    int lane = threadIdx.x & 31;
    if (wave >= N_NODES) return;
    const float* o = out + (size_t)wave * EMBED;
    float vx = o[2 * lane]     + bias[2 * lane];
    float vy = o[2 * lane + 1] + bias[2 * lane + 1];
    float ss = vx * vx + vy * vy;
#pragma unroll
    for (int off = 16; off > 0; off >>= 1) ss += __shfl_xor(ss, off, 32);
    float inv = 1.0f / fmaxf(sqrtf(ss), 1e-12f);
    float* t = total + (size_t)wave * EMBED;
    t[2 * lane]     += vx * inv;
    t[2 * lane + 1] += vy * inv;
}

// BPR: one wave per batch row; accumulate -log_sigmoid(s0-s1) into acc[0].
__global__ void k_bpr(const int* __restrict__ batch, const float* __restrict__ total,
                      float* __restrict__ acc, int bi) {
    int wave = ((int)blockIdx.x * blockDim.x + threadIdx.x) >> 5;
    int lane = threadIdx.x & 31;
    if (wave >= BATCH_N) return;
    const int* row = batch + (size_t)wave * 9 + (size_t)bi * 3;  // (B,3,3)
    int u = row[0], i0 = row[1], i1 = row[2];
    const float* uf = total + (size_t)u * EMBED;
    const float* f0 = total + ((size_t)N_USERS + i0) * EMBED;
    const float* f1 = total + ((size_t)N_USERS + i1) * EMBED;
    float2 a = *(const float2*)(uf + 2 * lane);
    float2 p = *(const float2*)(f0 + 2 * lane);
    float2 n = *(const float2*)(f1 + 2 * lane);
    float s0 = a.x * p.x + a.y * p.y;
    float s1 = a.x * n.x + a.y * n.y;
#pragma unroll
    for (int off = 16; off > 0; off >>= 1) {
        s0 += __shfl_xor(s0, off, 32);
        s1 += __shfl_xor(s1, off, 32);
    }
    if (lane == 0) {
        float z = s0 - s1;
        // -log_sigmoid(z) = max(-z,0) + log1p(exp(-|z|))
        float nl = fmaxf(-z, 0.f) + log1pf(expf(-fabsf(z)));
        atomicAdd(acc, nl);
    }
}

__global__ void k_sumsq(const float* __restrict__ p, long long n, float* __restrict__ acc) {
    float s = 0.f;
    long long i = (long long)blockIdx.x * blockDim.x + threadIdx.x;
    long long stride = (long long)gridDim.x * blockDim.x;
    for (; i < n; i += stride) { float v = p[i]; s += v * v; }
#pragma unroll
    for (int off = 16; off > 0; off >>= 1) s += __shfl_xor(s, off, 32);
    __shared__ float red[8];
    int lane = threadIdx.x & 31, w = threadIdx.x >> 5;
    if (lane == 0) red[w] = s;
    __syncthreads();
    if (threadIdx.x < 8) {
        s = red[threadIdx.x];
#pragma unroll
        for (int off = 4; off > 0; off >>= 1) s += __shfl_xor(s, off, 32);
        if (threadIdx.x == 0) atomicAdd(acc, s);
    }
}

__global__ void k_finish(const float* __restrict__ acc, float* __restrict__ out) {
    if (threadIdx.x == 0)
        out[0] = acc[0] / (float)BATCH_N + REG_W * 0.5f * acc[1] / (float)BATCH_N;
}

// ---------------- host launch ----------------

extern "C" void kernel_launch(void* const* d_in, const int* in_sizes, int n_in,
                              void* d_out, int out_size, void* d_ws, size_t ws_size,
                              hipStream_t stream) {
    const float* user  = (const float*)d_in[0];   // (100000,64)
    const float* item  = (const float*)d_in[1];   // (100000,64)
    const float* gw    = (const float*)d_in[2];   // (3,64,64)
    const float* gb    = (const float*)d_in[3];   // (3,64)
    const int*   edges = (const int*)d_in[4];     // (3,2,2000000)
    const int*   batch = (const int*)d_in[5];     // (4096,3,3)
    float* out = (float*)d_out;

    float* total = (float*)d_ws;                          // 12.8M f32
    float* x     = total + (size_t)N_NODES * EMBED;       // 12.8M f32
    float* gout  = x     + (size_t)N_NODES * EMBED;       // 12.8M f32
    float* deg   = gout  + (size_t)N_NODES * EMBED;       // 200k  (contiguous after gout)
    float* dinv  = deg + N_NODES;                         // 200k
    float* acc   = dinv + N_NODES;                        // 16 f32 (acc[0]=bpr sum, acc[1]=reg sum)
    _Float16* bp = (_Float16*)(acc + 16);                 // 4096 halves, 32B-aligned

    k_zero_acc<<<1, 32, 0, stream>>>(acc);
    k_init_total<<<12500, 256, 0, stream>>>(user, item, total);

    for (int bi = 0; bi < 3; ++bi) {
        const int* rows = edges + (size_t)bi * 2 * NEDGE;
        const int* cols = rows + NEDGE;
        k_pack_w<<<1, 256, 0, stream>>>(gw + (size_t)bi * EMBED * EMBED, bp);
        // zero gout + deg in one contiguous pass (13,000,000 floats)
        k_zero<<<8192, 256, 0, stream>>>(gout, (long long)(N_NODES * (EMBED + 1)) / 4);
        k_gemm<<<6250, 256, 0, stream>>>(total, bp, x);             // 50000 waves
        k_deg<<<(NEDGE + 255) / 256, 256, 0, stream>>>(cols, deg);
        k_dinv<<<(N_NODES + 255) / 256, 256, 0, stream>>>(deg, dinv);
        k_scatter<<<(NEDGE * 32) / 256, 256, 0, stream>>>(rows, cols, dinv, x, gout);
        k_norm_add<<<(N_NODES * 32) / 256, 256, 0, stream>>>(gout, gb + bi * EMBED, total);
        k_bpr<<<(BATCH_N * 32) / 256, 256, 0, stream>>>(batch, total, acc, bi);
    }

    k_sumsq<<<1024, 256, 0, stream>>>(user, (long long)N_USERS * EMBED, acc + 1);
    k_sumsq<<<1024, 256, 0, stream>>>(item, (long long)N_ITEMS * EMBED, acc + 1);
    k_finish<<<1, 32, 0, stream>>>(acc, out);
}
